// MoEFFN_83562883711460
// MI455X (gfx1250) — matrix operand order repaired and checked by reading, not compiled, not run on hardware
//
#include <hip/hip_runtime.h>

// ---------------------------------------------------------------------------
// MoE FFN (top-2 of 8 experts), MI455X / gfx1250, wave32 + WMMA bf16 + TDM.
// init -> weight fp32->bf16 (weights then L2-resident, 134MB < 192MB) ->
// router (softmax/top2, compacted per-expert token lists) -> scan ->
// GEMM1 (x@w1^T,+b1,gelu -> bf16 h) -> GEMM2 (h@w2^T,+b2,*combine, atomics).
// TDM tile staging is double-buffered: tensor_load_to_lds for chunk k+1 is in
// flight while chunk k runs 16x v_wmma_f32_16x16x32_bf16 (TENSORcnt pipelining).
// ---------------------------------------------------------------------------

typedef __attribute__((ext_vector_type(4)))  __bf16 v4bf;
typedef __attribute__((ext_vector_type(8)))  __bf16 v8bf;
typedef __attribute__((ext_vector_type(16))) __bf16 v16bf;
typedef __attribute__((ext_vector_type(8)))  float  v8f;
typedef __attribute__((ext_vector_type(4)))  unsigned int u32x4;
typedef __attribute__((ext_vector_type(8)))  int i32x8;
typedef __attribute__((ext_vector_type(4)))  int i32x4;

#define T_TOKENS 8192
#define C_DIM    1024
#define H_DIM    4096
#define N_EXP    8
#define BK       64

#if defined(__AMDGCN__) && __has_builtin(__builtin_amdgcn_tensor_load_to_lds)
#define HAVE_TDM 1
#else
#define HAVE_TDM 0
#endif

// --- TDM: 2D tile (tile_rows x tile_k bf16 elements) global -> LDS ----------
#if HAVE_TDM
__device__ __forceinline__ void tdm_load_2d(unsigned lds_off, const void* gptr,
                                            unsigned tile_k, unsigned tile_rows,
                                            unsigned long long row_stride_elems) {
  unsigned long long ga = (unsigned long long)(size_t)gptr;
  const unsigned td0 = 1u << 30, td1 = 1u << 30;   // huge tensor dims: no OOB clip
  u32x4 g0;
  g0[0] = 1u;                                       // count=1, user descriptor
  g0[1] = lds_off;                                  // lds_addr (bytes)
  g0[2] = (unsigned)(ga & 0xffffffffull);           // global_addr[31:0]
  g0[3] = (unsigned)((ga >> 32) & 0x01ffffffull) | (2u << 30);  // addr[56:32]|type=2
  i32x8 g1;
  g1[0] = (int)(1u << 16);                          // wg_mask=0, data_size=1 (2B)
  g1[1] = (int)((td0 & 0xffffu) << 16);             // atomic_addr=0 | dim0[15:0]
  g1[2] = (int)((td0 >> 16) & 0xffffu) | (int)((td1 & 0xffffu) << 16);
  g1[3] = (int)((td1 >> 16) & 0xffffu) | (int)(tile_k << 16);   // tile_dim0
  g1[4] = (int)(tile_rows & 0xffffu);               // tile_dim1 | tile_dim2=0
  g1[5] = (int)(row_stride_elems & 0xffffffffull);  // dim0_stride[31:0]
  g1[6] = (int)((row_stride_elems >> 32) & 0xffffull);
  g1[7] = 0;
  i32x4 z4 = {0, 0, 0, 0};
#if __clang_major__ >= 23
  i32x8 z8 = {0, 0, 0, 0, 0, 0, 0, 0};
  __builtin_amdgcn_tensor_load_to_lds(g0, g1, z4, z4, z8, 0);
#else
  __builtin_amdgcn_tensor_load_to_lds(g0, g1, z4, z4, 0);
#endif
}
#endif

// --- fragment loaders (contiguous k = 16*(lane/16) + s, same for A and B) ----------------------------
__device__ __forceinline__ v16bf ld_b_frag(const __bf16* p) {
  v8bf lo = *(const v8bf*)p;
  v8bf hi = *(const v8bf*)(p + 8);
  v16bf r;
#pragma unroll
  for (int i = 0; i < 8; ++i) { r[i] = lo[i]; r[i + 8] = hi[i]; }
  return r;
}
__device__ __forceinline__ v16bf ld_a_frag(const __bf16* row, int half) {
  v8bf lo = *(const v8bf*)(row + half * 16);
  v8bf hi = *(const v8bf*)(row + half * 16 + 8);
  v16bf r;
#pragma unroll
  for (int i = 0; i < 8; ++i) { r[i] = lo[i]; r[i + 8] = hi[i]; }
  return r;
}
__device__ __forceinline__ float fast_tanh(float v) {
#if __has_builtin(__builtin_amdgcn_tanhf)
  return __builtin_amdgcn_tanhf(v);           // v_tanh_f32 (TRANS)
#elif __has_builtin(__builtin_amdgcn_tanh_f32)
  return __builtin_amdgcn_tanh_f32(v);
#else
  return tanhf(v);
#endif
}
__device__ __forceinline__ float gelu_tanh(float v) {
  return 0.5f * v * (1.f + fast_tanh(0.7978845608028654f * (v + 0.044715f * v * v * v)));
}
__device__ __forceinline__ v8bf cvt8(const float4 a, const float4 b) {
  v8bf r;
  r[0] = (__bf16)a.x; r[1] = (__bf16)a.y; r[2] = (__bf16)a.z; r[3] = (__bf16)a.w;
  r[4] = (__bf16)b.x; r[5] = (__bf16)b.y; r[6] = (__bf16)b.z; r[7] = (__bf16)b.w;
  return r;
}
// gather-row fp32 -> bf16 staging (GEMM1 A operand)
__device__ __forceinline__ void stage_a32(__bf16* dst, const float* src) {
#pragma unroll
  for (int q = 0; q < 2; ++q) {
    float4 f0 = *(const float4*)(src + q * 16);
    float4 f1 = *(const float4*)(src + q * 16 + 4);
    float4 f2 = *(const float4*)(src + q * 16 + 8);
    float4 f3 = *(const float4*)(src + q * 16 + 12);
    *(v8bf*)(dst + q * 16)     = cvt8(f0, f1);
    *(v8bf*)(dst + q * 16 + 8) = cvt8(f2, f3);
  }
}

// --- zero output + routing state --------------------------------------------
__global__ __launch_bounds__(256)
void k_init(float* __restrict__ out, int* __restrict__ counts,
            int* __restrict__ offsets) {
  size_t i = (size_t)blockIdx.x * blockDim.x + threadIdx.x;
  ((float4*)out)[i] = make_float4(0.f, 0.f, 0.f, 0.f);
  if (blockIdx.x == 0 && threadIdx.x < N_EXP)     counts[threadIdx.x]  = 0;
  if (blockIdx.x == 0 && threadIdx.x < N_EXP + 1) offsets[threadIdx.x] = 0;
}

// --- weights fp32 -> bf16 ----------------------------------------------------
__global__ __launch_bounds__(256)
void k_convert(const float* __restrict__ w1, const float* __restrict__ w2,
               __bf16* __restrict__ w1b, __bf16* __restrict__ w2b) {
  const size_t n4 = (size_t)N_EXP * H_DIM * C_DIM / 4;
  const size_t stride = (size_t)gridDim.x * blockDim.x;
  for (size_t i = (size_t)blockIdx.x * blockDim.x + threadIdx.x; i < n4; i += stride) {
    float4 a = ((const float4*)w1)[i];
    v4bf r; r[0] = (__bf16)a.x; r[1] = (__bf16)a.y; r[2] = (__bf16)a.z; r[3] = (__bf16)a.w;
    ((v4bf*)w1b)[i] = r;
    float4 b = ((const float4*)w2)[i];
    v4bf s; s[0] = (__bf16)b.x; s[1] = (__bf16)b.y; s[2] = (__bf16)b.z; s[3] = (__bf16)b.w;
    ((v4bf*)w2b)[i] = s;
  }
}

// --- router: one wave32 per token -------------------------------------------
__global__ __launch_bounds__(256)
void k_router(const float* __restrict__ x, const float* __restrict__ gw,
              float* __restrict__ combine, int* __restrict__ counts,
              int* __restrict__ tok) {
  __shared__ float g[N_EXP * C_DIM];
  for (int i = threadIdx.x; i < N_EXP * C_DIM; i += 256) g[i] = gw[i];
  __syncthreads();
  const int lane = threadIdx.x & 31;
  const int wv   = threadIdx.x >> 5;
  const int t    = blockIdx.x * 8 + wv;
  const float* xr = x + (size_t)t * C_DIM;
  float acc[N_EXP];
#pragma unroll
  for (int e = 0; e < N_EXP; ++e) acc[e] = 0.f;
  for (int c = lane; c < C_DIM; c += 32) {
    float xv = xr[c];
#pragma unroll
    for (int e = 0; e < N_EXP; ++e) acc[e] = fmaf(xv, g[e * C_DIM + c], acc[e]);
  }
#pragma unroll
  for (int e = 0; e < N_EXP; ++e) {
#pragma unroll
    for (int off = 16; off > 0; off >>= 1) acc[e] += __shfl_xor(acc[e], off, 32);
  }
  if (lane == 0) {
    float mx = acc[0];
#pragma unroll
    for (int e = 1; e < N_EXP; ++e) mx = fmaxf(mx, acc[e]);
    float p[N_EXP], sum = 0.f;
#pragma unroll
    for (int e = 0; e < N_EXP; ++e) { p[e] = __expf(acc[e] - mx); sum += p[e]; }
    const float inv = 1.f / sum;
#pragma unroll
    for (int e = 0; e < N_EXP; ++e) p[e] *= inv;
    int i0 = 0;
#pragma unroll
    for (int e = 1; e < N_EXP; ++e) if (p[e] > p[i0]) i0 = e;
    int i1 = (i0 == 0) ? 1 : 0;
#pragma unroll
    for (int e = 0; e < N_EXP; ++e) if (e != i0 && p[e] > p[i1]) i1 = e;
    const float s  = p[i0] + p[i1] + 1e-8f;
    const float w0 = p[i0] / s, w1v = p[i1] / s;
#pragma unroll
    for (int e = 0; e < N_EXP; ++e) combine[t * N_EXP + e] = 0.f;
    combine[t * N_EXP + i0] = w0;
    combine[t * N_EXP + i1] = w1v;
    int s0 = atomicAdd(&counts[i0], 1); tok[i0 * T_TOKENS + s0] = t;
    int s1 = atomicAdd(&counts[i1], 1); tok[i1 * T_TOKENS + s1] = t;
  }
}

// --- exclusive scan over 8 expert counts ------------------------------------
__global__ void k_scan(const int* __restrict__ counts, int* __restrict__ offsets) {
  if (threadIdx.x == 0 && blockIdx.x == 0) {
    int s = 0;
#pragma unroll
    for (int e = 0; e < N_EXP; ++e) {
      offsets[e] = s;
      int c = counts[e]; if (c > T_TOKENS) c = T_TOKENS;
      s += c;
    }
    offsets[N_EXP] = s;
  }
}

// --- GEMM1: h = gelu(x_gathered @ w1^T + b1), K = C = 1024 ------------------
__global__ __launch_bounds__(256)
void k_gemm1(const float* __restrict__ x, const __bf16* __restrict__ w1b,
             const float* __restrict__ b1, const int* __restrict__ counts,
             const int* __restrict__ offsets, const int* __restrict__ tok,
             __bf16* __restrict__ hbuf) {
  const int e = blockIdx.z;
  int cnt = counts[e]; if (cnt > T_TOKENS) cnt = T_TOKENS;
  const int m0 = blockIdx.x * 128;
  if (m0 >= cnt) return;
  const int n0   = blockIdx.y * 128;
  const int eoff = offsets[e];
  __shared__ __bf16 As[2][128 * BK];
  __shared__ __bf16 Bs[2][128 * BK];
  const int tid = threadIdx.x;
  const int lane = tid & 31, wv = tid >> 5;
  const int wm = (wv & 1) * 64, wn = (wv >> 1) * 32;
  const int sr = tid >> 1, sh = tid & 1;       // staging: 2 threads / tile row
  int srow = m0 + sr; if (srow >= cnt) srow = cnt - 1;
  const int stok = tok[e * T_TOKENS + srow];
  const float*  asrc  = x + (size_t)stok * C_DIM + sh * 32;
  const __bf16* btile = w1b + ((size_t)e * H_DIM + n0) * C_DIM;   // [128 n][C k]
  const int sidx = sr * BK + sh * 32;

  v8f acc[4][2];
#pragma unroll
  for (int mt = 0; mt < 4; ++mt)
#pragma unroll
    for (int nt = 0; nt < 2; ++nt) acc[mt][nt] = 0.f;

  // prologue: stage chunk 0 into buffer 0
  stage_a32(&As[0][sidx], asrc);
#if HAVE_TDM
  if (wv == 0) tdm_load_2d((unsigned)(size_t)&Bs[0][0], btile, BK, 128,
                           (unsigned long long)C_DIM);
#else
  {
    const __bf16* bs = btile + (size_t)sr * C_DIM + sh * 32;
#pragma unroll
    for (int q = 0; q < 4; ++q) *(v8bf*)&Bs[0][sidx + q * 8] = *(const v8bf*)(bs + q * 8);
  }
#endif

  for (int k0 = 0; k0 < C_DIM; k0 += BK) {
    const int ib = (k0 >> 6) & 1;
    const int kn = k0 + BK;
    if (kn < C_DIM) {                          // issue next chunk into other buffer
      stage_a32(&As[ib ^ 1][sidx], asrc + kn);
      __builtin_prefetch(asrc + kn + BK, 0, 1);
#if HAVE_TDM
      if (wv == 0) {
        tdm_load_2d((unsigned)(size_t)&Bs[ib ^ 1][0], btile + kn, BK, 128,
                    (unsigned long long)C_DIM);
        __builtin_amdgcn_s_wait_tensorcnt(1);  // current B tile has landed
      }
#else
      {
        const __bf16* bs = btile + (size_t)sr * C_DIM + kn + sh * 32;
#pragma unroll
        for (int q = 0; q < 4; ++q)
          *(v8bf*)&Bs[ib ^ 1][sidx + q * 8] = *(const v8bf*)(bs + q * 8);
      }
#endif
    }
#if HAVE_TDM
    else if (wv == 0) {
      __builtin_amdgcn_s_wait_tensorcnt(0);
    }
#endif
    __syncthreads();                           // As[ib]/Bs[ib] visible to all
#pragma unroll
    for (int kk = 0; kk < BK; kk += 32) {
      v16bf af[4], bfr[2];
#pragma unroll
      for (int mt = 0; mt < 4; ++mt)
        af[mt] = ld_a_frag(&As[ib][(wm + mt * 16 + (lane & 15)) * BK + kk], lane >> 4);
#pragma unroll
      for (int nt = 0; nt < 2; ++nt)
        bfr[nt] = ld_b_frag(&Bs[ib][(wn + nt * 16 + (lane & 15)) * BK + kk + ((lane >> 4) << 4)]);
#pragma unroll
      for (int mt = 0; mt < 4; ++mt)
#pragma unroll
        for (int nt = 0; nt < 2; ++nt)
          acc[mt][nt] = __builtin_amdgcn_wmma_f32_16x16x32_bf16(
              false, af[mt], false, bfr[nt], (short)0, acc[mt][nt], false, false);
    }
    __syncthreads();                           // buffer ib free for restage
  }

  const int lr16 = (lane >> 4) << 3;
  const int lc   = lane & 15;
#pragma unroll
  for (int mt = 0; mt < 4; ++mt) {
#pragma unroll
    for (int nt = 0; nt < 2; ++nt) {
      const int   hcol = n0 + wn + nt * 16 + lc;
      const float bb   = b1[e * H_DIM + hcol];
#pragma unroll
      for (int i = 0; i < 8; ++i) {
        const int grow = m0 + wm + mt * 16 + lr16 + i;
        if (grow < cnt) {
          float v = acc[mt][nt][i] + bb;
          hbuf[((size_t)(eoff + grow)) * H_DIM + hcol] = (__bf16)gelu_tanh(v);
        }
      }
    }
  }
}

// --- GEMM2: out += combine * (h @ w2^T + b2), K = H = 4096 ------------------
__global__ __launch_bounds__(256)
void k_gemm2(const __bf16* __restrict__ hbuf, const __bf16* __restrict__ w2b,
             const float* __restrict__ b2, const int* __restrict__ counts,
             const int* __restrict__ offsets, const int* __restrict__ tok,
             const float* __restrict__ combine, float* __restrict__ out) {
  const int e = blockIdx.z;
  int cnt = counts[e]; if (cnt > T_TOKENS) cnt = T_TOKENS;
  const int m0 = blockIdx.x * 128;
  if (m0 >= cnt) return;
  const int n0   = blockIdx.y * 128;
  const int eoff = offsets[e];
  __shared__ __bf16 As[2][128 * BK];
  __shared__ __bf16 Bs[2][128 * BK];
  const int tid = threadIdx.x;
  const int lane = tid & 31, wv = tid >> 5;
  const int wm = (wv & 1) * 64, wn = (wv >> 1) * 32;
  const __bf16* atile = hbuf + ((size_t)(eoff + m0)) * H_DIM;     // contiguous rows
  const __bf16* btile = w2b + ((size_t)e * C_DIM + n0) * H_DIM;
#if !HAVE_TDM
  const int sr = tid >> 1, sh = tid & 1;
  int srow = m0 + sr; if (srow >= cnt) srow = cnt - 1;
  const __bf16* asrc = hbuf + ((size_t)(eoff + srow)) * H_DIM + sh * 32;
  const __bf16* bsrc = btile + (size_t)sr * H_DIM + sh * 32;
  const int sidx = sr * BK + sh * 32;
#endif

  v8f acc[4][2];
#pragma unroll
  for (int mt = 0; mt < 4; ++mt)
#pragma unroll
    for (int nt = 0; nt < 2; ++nt) acc[mt][nt] = 0.f;

  // prologue: dual TDM for chunk 0 into buffer 0
#if HAVE_TDM
  if (wv == 0) {
    tdm_load_2d((unsigned)(size_t)&As[0][0], atile, BK, 128, (unsigned long long)H_DIM);
    tdm_load_2d((unsigned)(size_t)&Bs[0][0], btile, BK, 128, (unsigned long long)H_DIM);
  }
#else
#pragma unroll
  for (int q = 0; q < 4; ++q) {
    *(v8bf*)&As[0][sidx + q * 8] = *(const v8bf*)(asrc + q * 8);
    *(v8bf*)&Bs[0][sidx + q * 8] = *(const v8bf*)(bsrc + q * 8);
  }
#endif

  for (int k0 = 0; k0 < H_DIM; k0 += BK) {
    const int ib = (k0 >> 6) & 1;
    const int kn = k0 + BK;
#if HAVE_TDM
    if (wv == 0) {
      if (kn < H_DIM) {                        // DMA next chunk while computing
        tdm_load_2d((unsigned)(size_t)&As[ib ^ 1][0], atile + kn, BK, 128,
                    (unsigned long long)H_DIM);
        tdm_load_2d((unsigned)(size_t)&Bs[ib ^ 1][0], btile + kn, BK, 128,
                    (unsigned long long)H_DIM);
        __builtin_amdgcn_s_wait_tensorcnt(2);  // current A+B pair has landed
      } else {
        __builtin_amdgcn_s_wait_tensorcnt(0);
      }
    }
#else
    if (kn < H_DIM) {
#pragma unroll
      for (int q = 0; q < 4; ++q) {
        *(v8bf*)&As[ib ^ 1][sidx + q * 8] = *(const v8bf*)(asrc + kn + q * 8);
        *(v8bf*)&Bs[ib ^ 1][sidx + q * 8] = *(const v8bf*)(bsrc + kn + q * 8);
      }
    }
#endif
    __syncthreads();
#pragma unroll
    for (int kk = 0; kk < BK; kk += 32) {
      v16bf af[4], bfr[2];
#pragma unroll
      for (int mt = 0; mt < 4; ++mt)
        af[mt] = ld_a_frag(&As[ib][(wm + mt * 16 + (lane & 15)) * BK + kk], lane >> 4);
#pragma unroll
      for (int nt = 0; nt < 2; ++nt)
        bfr[nt] = ld_b_frag(&Bs[ib][(wn + nt * 16 + (lane & 15)) * BK + kk + ((lane >> 4) << 4)]);
#pragma unroll
      for (int mt = 0; mt < 4; ++mt)
#pragma unroll
        for (int nt = 0; nt < 2; ++nt)
          acc[mt][nt] = __builtin_amdgcn_wmma_f32_16x16x32_bf16(
              false, af[mt], false, bfr[nt], (short)0, acc[mt][nt], false, false);
    }
    __syncthreads();
  }

  const int lr16 = (lane >> 4) << 3;
  const int lc   = lane & 15;
#pragma unroll
  for (int mt = 0; mt < 4; ++mt) {
#pragma unroll
    for (int i = 0; i < 8; ++i) {
      const int grow = m0 + wm + mt * 16 + lr16 + i;
      if (grow >= cnt) continue;
      const int   t     = tok[e * T_TOKENS + grow];
      const float scale = combine[t * N_EXP + e];
#pragma unroll
      for (int nt = 0; nt < 2; ++nt) {
        const int   ocol = n0 + wn + nt * 16 + lc;
        const float y    = acc[mt][nt][i] + b2[e * C_DIM + ocol];
        atomicAdd(&out[(size_t)t * C_DIM + ocol], scale * y);
      }
    }
  }
}

// ---------------------------------------------------------------------------
extern "C" void kernel_launch(void* const* d_in, const int* in_sizes, int n_in,
                              void* d_out, int out_size, void* d_ws, size_t ws_size,
                              hipStream_t stream) {
  (void)in_sizes; (void)n_in; (void)out_size; (void)ws_size;
  const float* x      = (const float*)d_in[0];
  const float* gate_w = (const float*)d_in[1];
  const float* w1     = (const float*)d_in[2];
  const float* b1     = (const float*)d_in[3];
  const float* w2     = (const float*)d_in[4];
  const float* b2     = (const float*)d_in[5];
  float* out = (float*)d_out;

  char* p = (char*)d_ws;
  __bf16* w1b = (__bf16*)p; p += (size_t)N_EXP * H_DIM * C_DIM * 2;   // 64 MB
  __bf16* w2b = (__bf16*)p; p += (size_t)N_EXP * H_DIM * C_DIM * 2;   // 64 MB
  __bf16* hbuf = (__bf16*)p; p += (size_t)T_TOKENS * 2 * H_DIM * 2;   // 128 MB
  float* combine = (float*)p; p += (size_t)T_TOKENS * N_EXP * 4;      // 256 KB
  int* counts  = (int*)p; p += 256;
  int* offsets = (int*)p; p += 256;
  int* tok     = (int*)p; p += (size_t)N_EXP * T_TOKENS * 4;          // 256 KB

  k_init<<<T_TOKENS * C_DIM / (256 * 4), 256, 0, stream>>>(out, counts, offsets);
  k_convert<<<4096, 256, 0, stream>>>(w1, w2, w1b, w2b);
  k_router<<<T_TOKENS / 8, 256, 0, stream>>>(x, gate_w, combine, counts, tok);
  k_scan<<<1, 32, 0, stream>>>(counts, offsets);
  k_gemm1<<<dim3(T_TOKENS / 128, H_DIM / 128, N_EXP), 256, 0, stream>>>(
      x, w1b, b1, counts, offsets, tok, hbuf);
  k_gemm2<<<dim3(T_TOKENS / 128, C_DIM / 128, N_EXP), 256, 0, stream>>>(
      hbuf, w2b, b2, counts, offsets, tok, combine, out);
}
